// _MultiHeadSelfAttention_41755672052412
// MI455X (gfx1250) — matrix operand order repaired and checked by reading, not compiled, not run on hardware
//
#include <hip/hip_runtime.h>
#include <hip/hip_bf16.h>

// ---------------------------------------------------------------------------
// MI455X (gfx1250) multi-head self-attention, bf16 WMMA path.
//   v_wmma_f32_16x16x32_bf16 + global_load_async_to_lds_b128 staging.
// ---------------------------------------------------------------------------

typedef __bf16 bf16_t;
typedef __attribute__((ext_vector_type(16))) __bf16 v16bf;
typedef __attribute__((ext_vector_type(8)))  float  v8f;

#define WMMA_BF16(a, b, c) \
  __builtin_amdgcn_wmma_f32_16x16x32_bf16(false, (a), false, (b), (short)0, (c), false, false)

// Async global->LDS DMA: copies 32B (two b128) per lane; INST_OFFSET applies to
// both the LDS destination and the global source, so one address pair suffices.
__device__ __forceinline__ void async_copy_32B(unsigned lds_addr, unsigned long long gaddr) {
  asm volatile("global_load_async_to_lds_b128 %0, %1, off\n\t"
               "global_load_async_to_lds_b128 %0, %1, off offset:16"
               :: "v"(lds_addr), "v"(gaddr) : "memory");
}
__device__ __forceinline__ void async_copy_64B(unsigned lds_addr, unsigned long long gaddr) {
  asm volatile("global_load_async_to_lds_b128 %0, %1, off\n\t"
               "global_load_async_to_lds_b128 %0, %1, off offset:16\n\t"
               "global_load_async_to_lds_b128 %0, %1, off offset:32\n\t"
               "global_load_async_to_lds_b128 %0, %1, off offset:48"
               :: "v"(lds_addr), "v"(gaddr) : "memory");
}
__device__ __forceinline__ void wait_async() {
  asm volatile("s_wait_asynccnt 0x0" ::: "memory");
}

static constexpr int kDModel = 1024;
static constexpr int kHeads  = 16;
static constexpr int kDHead  = 64;     // kDModel / kHeads
static constexpr int kSeq    = 2048;
static constexpr int kBatch  = 4;
static constexpr int kRows   = kBatch * kSeq;   // 8192

// ---------------------------------------------------------------------------
// f32 -> bf16 conversion (grid-stride)
// ---------------------------------------------------------------------------
__global__ void __launch_bounds__(256)
cvt_f32_bf16_kernel(const float* __restrict__ in, bf16_t* __restrict__ out, int n) {
  int i = blockIdx.x * blockDim.x + threadIdx.x;
  int stride = gridDim.x * blockDim.x;
  for (; i < n; i += stride) out[i] = (bf16_t)in[i];
}

// ---------------------------------------------------------------------------
// bf16 WMMA GEMM:  C[M,N] = A[M,K] * B[K,N] + bias[N]
//   mode 0: N = 3072, scatter into head-major Q/K/V (bf16) for attention
//   mode 1: N = 1024, write f32 to fout
// Block: 256 threads (8 waves), tile 128x128, wave tile 32x64, K-slab 64
// (two 16x16x32 WMMA k-steps per staging round). A-tile staged with async
// global->LDS DMA; B-tile transposed through VGPRs.
// ---------------------------------------------------------------------------
__global__ void __launch_bounds__(256)
gemm_bf16_wmma_kernel(const bf16_t* __restrict__ A, const bf16_t* __restrict__ B,
                      const float* __restrict__ bias, int M, int N, int Kd, int mode,
                      bf16_t* __restrict__ qout, bf16_t* __restrict__ kout,
                      bf16_t* __restrict__ vout, float* __restrict__ fout) {
  __shared__ bf16_t As[128][72];   // row-major A tile (64 k + pad), 144B row stride
  __shared__ bf16_t Bs[128][72];   // B tile transposed: [n][k], padded

  const int tid  = threadIdx.x;
  const int w    = tid >> 5;
  const int lane = tid & 31;
  const int ln   = lane & 15;
  const int half = lane >> 4;
  const int wm   = w & 3;          // 4 waves along M
  const int wn   = w >> 2;         // 2 waves along N
  const int m0   = blockIdx.x * 128;
  const int n0   = blockIdx.y * 128;

  const v8f vzero = {0.f, 0.f, 0.f, 0.f, 0.f, 0.f, 0.f, 0.f};
  v8f acc[2][4];
#pragma unroll
  for (int r = 0; r < 2; ++r)
#pragma unroll
    for (int c = 0; c < 4; ++c) acc[r][c] = vzero;

  const int arow = tid >> 1;            // 0..127
  const int ak   = (tid & 1) * 32;      // 0 or 32
  const int bk   = tid >> 2;            // 0..63
  const int bn   = (tid & 3) * 32;      // 0,32,64,96

  const int nk = Kd >> 6;               // K-slab = 64
  for (int kt = 0; kt < nk; ++kt) {
    if (kt + 1 < nk) {   // prefetch next K-slab -> global_prefetch_b8
      __builtin_prefetch(A + (size_t)(m0 + arow) * Kd + (kt + 1) * 64 + ak, 0, 0);
      __builtin_prefetch(B + (size_t)((kt + 1) * 64 + bk) * N + n0 + bn, 0, 0);
    }
    // stage A tile: 64B per lane via async global->LDS DMA (ASYNCcnt path)
    {
      const bf16_t* asrc = A + (size_t)(m0 + arow) * Kd + kt * 64 + ak;
      async_copy_64B((unsigned)(uintptr_t)&As[arow][ak],
                     (unsigned long long)(uintptr_t)asrc);
    }
    // stage B tile transposed (read 64B contiguous along N, scatter along k)
    {
      const bf16_t* bsrc = B + (size_t)(kt * 64 + bk) * N + n0 + bn;
      uint4 r0 = *reinterpret_cast<const uint4*>(bsrc);
      uint4 r1 = *reinterpret_cast<const uint4*>(bsrc + 8);
      uint4 r2 = *reinterpret_cast<const uint4*>(bsrc + 16);
      uint4 r3 = *reinterpret_cast<const uint4*>(bsrc + 24);
      alignas(16) bf16_t tb[32];
      *reinterpret_cast<uint4*>(&tb[0])  = r0;
      *reinterpret_cast<uint4*>(&tb[8])  = r1;
      *reinterpret_cast<uint4*>(&tb[16]) = r2;
      *reinterpret_cast<uint4*>(&tb[24]) = r3;
#pragma unroll
      for (int i = 0; i < 32; ++i) Bs[bn + i][bk] = tb[i];
    }
    wait_async();
    __syncthreads();

    // two WMMA k-steps per staged slab
#pragma unroll
    for (int ks = 0; ks < 2; ++ks) {
      v16bf afr[2], bfr[4];
#pragma unroll
      for (int r = 0; r < 2; ++r) {
        uint4* ap = reinterpret_cast<uint4*>(&afr[r]);
        const bf16_t* arp = &As[wm * 32 + r * 16 + ln][ks * 32];
        ap[0] = *reinterpret_cast<const uint4*>(arp + half * 8);        // K 0..7 | 8..15
        ap[1] = *reinterpret_cast<const uint4*>(arp + 16 + half * 8);   // K 16..23 | 24..31
      }
#pragma unroll
      for (int c = 0; c < 4; ++c) {
        uint4* bp = reinterpret_cast<uint4*>(&bfr[c]);
        const bf16_t* bcp = &Bs[wn * 64 + c * 16 + ln][ks * 32];
        bp[0] = *reinterpret_cast<const uint4*>(bcp + half * 8);        // K 0..7 | 8..15
        bp[1] = *reinterpret_cast<const uint4*>(bcp + 16 + half * 8);
      }
#pragma unroll
      for (int r = 0; r < 2; ++r)
#pragma unroll
        for (int c = 0; c < 4; ++c)
          acc[r][c] = WMMA_BF16(afr[r], bfr[c], acc[r][c]);
    }
    __syncthreads();
  }

  // epilogue: C layout M = jj + 8*half (per VGPR jj), N = lane&15 within frag
#pragma unroll
  for (int r = 0; r < 2; ++r)
#pragma unroll
    for (int c = 0; c < 4; ++c)
#pragma unroll
      for (int jj = 0; jj < 8; ++jj) {
        int m = m0 + wm * 32 + r * 16 + jj + 8 * half;
        int n = n0 + wn * 64 + c * 16 + ln;
        float val = acc[r][c][jj] + bias[n];
        if (mode == 0) {
          int sel = n >> 10;      // 0=Q 1=K 2=V
          int d   = n & 1023;
          int h   = d >> 6;
          int dh  = d & 63;
          int bb  = m >> 11;
          int s   = m & 2047;
          size_t idx = (((size_t)(bb * kHeads + h) * kSeq) + s) * kDHead + dh;
          bf16_t bv = (bf16_t)val;
          if (sel == 0)      qout[idx] = bv;
          else if (sel == 1) kout[idx] = bv;
          else               vout[idx] = bv;
        } else {
          fout[(size_t)m * kDModel + n] = val;
        }
      }
}

// ---------------------------------------------------------------------------
// Flash attention over head-major bf16 Q/K/V [bh, seq, 64].
// Block = (128 q-rows, one bh); 8 waves, each owns a 16-row q strip.
// Key blocks of 64; K staged via async DMA; online softmax; P re-layout
// C->A through padded LDS. Output written interleaved [b, s, d_model] bf16.
// ---------------------------------------------------------------------------
__global__ void __launch_bounds__(256)
attn_flash_kernel(const bf16_t* __restrict__ Q, const bf16_t* __restrict__ K,
                  const bf16_t* __restrict__ V, bf16_t* __restrict__ O) {
  __shared__ bf16_t Ks[64][72];       // [key][d]  (serves directly as K^T B-frags)
  __shared__ bf16_t Vs[64][72];       // [d][key]  (transposed for PV B-frags)
  __shared__ bf16_t Ps[8][16][72];    // per-wave P strip, padded rows

  const int tid  = threadIdx.x;
  const int w    = tid >> 5;
  const int lane = tid & 31;
  const int ln   = lane & 15;
  const int half = lane >> 4;

  const int bh = blockIdx.y;
  const int b  = bh >> 4;
  const int h  = bh & 15;
  const int q0 = blockIdx.x * 128 + w * 16;

  const bf16_t* Qh = Q + (size_t)bh * kSeq * kDHead;
  const bf16_t* Kh = K + (size_t)bh * kSeq * kDHead;
  const bf16_t* Vh = V + (size_t)bh * kSeq * kDHead;

  // preload Q strip as two A-fragments (K-dim = d_head, 2 steps of 32)
  v16bf qf[2];
  {
    const bf16_t* qptr = Qh + (size_t)(q0 + ln) * kDHead;
#pragma unroll
    for (int kk = 0; kk < 2; ++kk) {
      uint4* qp = reinterpret_cast<uint4*>(&qf[kk]);
      qp[0] = *reinterpret_cast<const uint4*>(qptr + kk * 32 + half * 8);
      qp[1] = *reinterpret_cast<const uint4*>(qptr + kk * 32 + 16 + half * 8);
    }
  }

  const v8f vzero = {0.f, 0.f, 0.f, 0.f, 0.f, 0.f, 0.f, 0.f};
  v8f oacc[4];
#pragma unroll
  for (int f = 0; f < 4; ++f) oacc[f] = vzero;
  float mrow[8], lrow[8];
#pragma unroll
  for (int jj = 0; jj < 8; ++jj) { mrow[jj] = -1e30f; lrow[jj] = 0.f; }

  const int skey = tid >> 2;          // 0..63
  const int sd0  = (tid & 3) * 16;    // 0,16,32,48

  for (int j = 0; j < kSeq / 64; ++j) {
    // ---- stage K via async DMA, V via VGPR transpose ----
    {
      const bf16_t* ksrc = Kh + (size_t)(j * 64 + skey) * kDHead + sd0;
      async_copy_32B((unsigned)(uintptr_t)&Ks[skey][sd0],
                     (unsigned long long)(uintptr_t)ksrc);

      const bf16_t* vsrc = Vh + (size_t)(j * 64 + skey) * kDHead + sd0;
      uint4 v0 = *reinterpret_cast<const uint4*>(vsrc);
      uint4 v1 = *reinterpret_cast<const uint4*>(vsrc + 8);
      alignas(16) bf16_t tv[16];
      *reinterpret_cast<uint4*>(&tv[0]) = v0;
      *reinterpret_cast<uint4*>(&tv[8]) = v1;
#pragma unroll
      for (int i = 0; i < 16; ++i) Vs[sd0 + i][skey] = tv[i];
    }
    wait_async();
    __syncthreads();

    // ---- S = Q * K^T  (16 x 64 per wave; 8 WMMAs) ----
    v8f sc[4];
#pragma unroll
    for (int c = 0; c < 4; ++c) sc[c] = vzero;
#pragma unroll
    for (int kk = 0; kk < 2; ++kk) {
#pragma unroll
      for (int c = 0; c < 4; ++c) {
        v16bf bkf;
        uint4* bp = reinterpret_cast<uint4*>(&bkf);
        const bf16_t* src = &Ks[c * 16 + ln][kk * 32 + half * 8];
        bp[0] = *reinterpret_cast<const uint4*>(src);
        bp[1] = *reinterpret_cast<const uint4*>(src + 16);
        sc[c] = WMMA_BF16(qf[kk], bkf, sc[c]);
      }
    }

    // ---- online softmax (row = jj + 8*half, replicated over 16-lane group) ----
#pragma unroll
    for (int jj = 0; jj < 8; ++jj) {
      float mx = -1e30f;
#pragma unroll
      for (int c = 0; c < 4; ++c) {
        float t = sc[c][jj] * 0.125f;   // 1/sqrt(64)
        sc[c][jj] = t;
        mx = fmaxf(mx, t);
      }
#pragma unroll
      for (int off = 1; off < 16; off <<= 1) mx = fmaxf(mx, __shfl_xor(mx, off, 32));
      float mnew  = fmaxf(mrow[jj], mx);
      float alpha = __expf(mrow[jj] - mnew);
      float lsum  = 0.f;
#pragma unroll
      for (int c = 0; c < 4; ++c) {
        float p = __expf(sc[c][jj] - mnew);
        sc[c][jj] = p;
        lsum += p;
      }
#pragma unroll
      for (int off = 1; off < 16; off <<= 1) lsum += __shfl_xor(lsum, off, 32);
      lrow[jj] = lrow[jj] * alpha + lsum;
      mrow[jj] = mnew;
#pragma unroll
      for (int f = 0; f < 4; ++f) oacc[f][jj] *= alpha;
    }

    // ---- write P (C layout) into per-wave LDS strip ----
#pragma unroll
    for (int c = 0; c < 4; ++c)
#pragma unroll
      for (int jj = 0; jj < 8; ++jj)
        Ps[w][jj + 8 * half][c * 16 + ln] = (bf16_t)sc[c][jj];
    __syncthreads();

    // ---- O += P * V  (A from Ps, B from Vs; 8 WMMAs) ----
#pragma unroll
    for (int kk = 0; kk < 2; ++kk) {
      v16bf pa;
      uint4* pp = reinterpret_cast<uint4*>(&pa);
      const bf16_t* prow = &Ps[w][ln][0];
      pp[0] = *reinterpret_cast<const uint4*>(prow + kk * 32 + half * 8);
      pp[1] = *reinterpret_cast<const uint4*>(prow + kk * 32 + 16 + half * 8);
#pragma unroll
      for (int f = 0; f < 4; ++f) {
        v16bf bvf;
        uint4* bp = reinterpret_cast<uint4*>(&bvf);
        const bf16_t* vrow = &Vs[f * 16 + ln][kk * 32 + half * 8];
        bp[0] = *reinterpret_cast<const uint4*>(vrow);
        bp[1] = *reinterpret_cast<const uint4*>(vrow + 16);
        oacc[f] = WMMA_BF16(pa, bvf, oacc[f]);
      }
    }
    __syncthreads();
  }

  // ---- normalize and write back interleaved [b, s, d_model] ----
#pragma unroll
  for (int f = 0; f < 4; ++f)
#pragma unroll
    for (int jj = 0; jj < 8; ++jj) {
      int srow = q0 + jj + 8 * half;
      int d    = f * 16 + ln;
      float val = oacc[f][jj] / lrow[jj];
      O[((size_t)(b * kSeq + srow)) * kDModel + h * kDHead + d] = (bf16_t)val;
    }
}

// ---------------------------------------------------------------------------
// Host-side orchestration
// ---------------------------------------------------------------------------
extern "C" void kernel_launch(void* const* d_in, const int* in_sizes, int n_in,
                              void* d_out, int out_size, void* d_ws, size_t ws_size,
                              hipStream_t stream) {
  (void)in_sizes; (void)n_in; (void)out_size; (void)ws_size;

  const float* x     = (const float*)d_in[0];   // [4,2048,1024]
  const float* w_qkv = (const float*)d_in[1];   // [1024,3072]
  const float* b_qkv = (const float*)d_in[2];   // [3072]
  const float* w_out = (const float*)d_in[3];   // [1024,1024]
  const float* b_out = (const float*)d_in[4];   // [1024]
  float* out = (float*)d_out;                   // [4,2048,1024] f32

  char* ws = (char*)d_ws;
  size_t off = 0;
  auto alloc = [&](size_t bytes) -> void* {
    void* p = ws + off;
    off += (bytes + 255) & ~(size_t)255;
    return p;
  };
  bf16_t* xb    = (bf16_t*)alloc((size_t)kRows * kDModel * 2);       // 16 MB
  bf16_t* wqkvb = (bf16_t*)alloc((size_t)kDModel * 3 * kDModel * 2); // 6 MB
  bf16_t* wob   = (bf16_t*)alloc((size_t)kDModel * kDModel * 2);     // 2 MB
  bf16_t* qb    = (bf16_t*)alloc((size_t)kRows * kDModel * 2);       // 16 MB
  bf16_t* kb    = (bf16_t*)alloc((size_t)kRows * kDModel * 2);       // 16 MB
  bf16_t* vb    = (bf16_t*)alloc((size_t)kRows * kDModel * 2);       // 16 MB
  bf16_t* aob   = (bf16_t*)alloc((size_t)kRows * kDModel * 2);       // 16 MB

  cvt_f32_bf16_kernel<<<2048, 256, 0, stream>>>(x, xb, kRows * kDModel);
  cvt_f32_bf16_kernel<<<2048, 256, 0, stream>>>(w_qkv, wqkvb, kDModel * 3 * kDModel);
  cvt_f32_bf16_kernel<<<1024, 256, 0, stream>>>(w_out, wob, kDModel * kDModel);

  // QKV projection: [8192,1024] x [1024,3072] -> head-major Q/K/V (bf16, bias folded)
  gemm_bf16_wmma_kernel<<<dim3(kRows / 128, (3 * kDModel) / 128), 256, 0, stream>>>(
      xb, wqkvb, b_qkv, kRows, 3 * kDModel, kDModel, /*mode=*/0, qb, kb, vb, nullptr);

  // Flash attention per (128 q-rows, bh)
  attn_flash_kernel<<<dim3(kSeq / 128, kBatch * kHeads), 256, 0, stream>>>(qb, kb, vb, aob);

  // Output projection: [8192,1024] x [1024,1024] + b_out -> f32 d_out
  gemm_bf16_wmma_kernel<<<dim3(kRows / 128, kDModel / 128), 256, 0, stream>>>(
      aob, wob, b_out, kRows, kDModel, kDModel, /*mode=*/1, nullptr, nullptr, nullptr, out);
}
